// XlaQuantizedLinear_30837865185753
// MI455X (gfx1250) — matrix operand order repaired and checked by reading, not compiled, not run on hardware
//
#include <hip/hip_runtime.h>
#include <stdint.h>

// ---------------------------------------------------------------------------
// Quantized matmul: out[b,o] = (sum_i x[b,i] * w_int8[o,i]) * scaler[o]
// B=16, IN=8192, OUT=28672.
//
// MI455X roofline: W is 235 MB int8 streamed once -> ~10 us @ 23.3 TB/s;
// 7.5 GFLOP -> purely bandwidth bound. f16 WMMA (int8 exact in f16),
// perm+xor magic dequant (2 VALU / 2 weights), +1152 bias folded out via
// precomputed activation row sums. Weights load NON-TEMPORAL (235 MB > 192 MB
// L2, stream-once) so they don't evict the hot 256 KB repacked activations.
// Each wave computes TWO 16-wide N tiles sharing one A fragment: halves the
// A-load instruction count and doubles accumulator ILP across load waits.
// ---------------------------------------------------------------------------

typedef _Float16 v16h __attribute__((ext_vector_type(16)));
typedef float    v8f  __attribute__((ext_vector_type(8)));
typedef uint32_t v4u  __attribute__((ext_vector_type(4)));

#define B_DIM   16
#define IN_DIM  8192
#define OUT_DIM 28672
#define NTILES  2                    // N tiles per wave (share A fragment)
#define KSPLIT  4                    // waves per block splitting K
#define KCHUNK  (IN_DIM / KSPLIT)    // 2048
#define KSTEPS  (KCHUNK / 32)        // 64 wmma steps per wave

// ---------------------------------------------------------------------------
// Prepass 1: x[16][8192] f32 -> f16, pre-swizzled into WMMA A-fragment order.
// A order used here (16-bit A 16x32, contiguous k, same as B): lane L<16 holds M=L, K={0..15};
// lane L+16 holds M=L, K={16..31}; element j of the lane's 16 halfs:
//   k_local = j + 16*hi,   hi = L>>4
// Stored so lane L's 16 halfs for k-block kb are contiguous at (kb*32+L)*16.
// ---------------------------------------------------------------------------
__global__ void xa_repack_kernel(const float* __restrict__ x,
                                 _Float16* __restrict__ xh) {
    int t   = blockIdx.x * blockDim.x + threadIdx.x;   // 0 .. 131071
    int kb  = t >> 9;                                  // k-block (32 wide)
    int rem = t & 511;
    int L   = rem >> 4;                                // lane 0..31
    int j   = rem & 15;                                // element in fragment
    int m   = L & 15;
    int hi  = L >> 4;
    int kl  = j + 16 * hi;
    xh[t] = (_Float16)x[m * IN_DIM + kb * 32 + kl];
}

// ---------------------------------------------------------------------------
// Prepass 2: rowsum[m] = sum_k (f32)(f16)x[m,k]  (matches values fed to WMMA)
// ---------------------------------------------------------------------------
__global__ void xa_rowsum_kernel(const float* __restrict__ x,
                                 float* __restrict__ rs) {
    int row  = threadIdx.x >> 5;
    int lane = threadIdx.x & 31;
    float s = 0.0f;
    for (int i = lane; i < IN_DIM; i += 32)
        s += (float)(_Float16)x[row * IN_DIM + i];
    for (int off = 16; off > 0; off >>= 1)
        s += __shfl_xor(s, off, 32);
    if (lane == 0) rs[row] = s;
}

// int8x4 dword -> two packed-f16 dwords holding (w + 1152); see header comment.
__device__ __forceinline__ void dq4(uint32_t d, uint32_t* lo, uint32_t* hi) {
    *lo = __builtin_amdgcn_perm(0u, d, 0x04010400u) ^ 0x64806480u;
    *hi = __builtin_amdgcn_perm(0u, d, 0x04030402u) ^ 0x64806480u;
}

// ---------------------------------------------------------------------------
// Main kernel: one block (4 waves) per 32-wide N tile pair; waves split K 4x.
// B order used here (32x16, same k order as A): lane = column n (mod 16), lane-half selects
// K half (0..15 / 16..31), VGPR v holds k = 16*hi + {2v,2v+1}. So each lane
// loads 16 contiguous int8 weights per tile: row (o0+n), bytes k0+16*hi..+15.
// ---------------------------------------------------------------------------
__global__ void __launch_bounds__(KSPLIT * 32)
qmm_wmma_kernel(const _Float16* __restrict__ xh,
                const int8_t*   __restrict__ w,
                const float*    __restrict__ scaler,
                const float*    __restrict__ rowsum,
                float*          __restrict__ out) {
    __shared__ float lds[KSPLIT][NTILES * 256];

    const int tid  = threadIdx.x;
    const int wave = tid >> 5;
    const int lane = tid & 31;
    const int o0   = blockIdx.x * (NTILES * 16);
    const int n    = lane & 15;
    const int hi   = lane >> 4;

    const int kb0 = wave * KSTEPS;                 // first 32-wide k-block
    const int8_t*   wp0 = w + (size_t)(o0 + n) * IN_DIM + (size_t)kb0 * 32 + hi * 16;
    const int8_t*   wp1 = wp0 + (size_t)16 * IN_DIM;       // second N tile
    const _Float16* ap  = xh + (size_t)kb0 * 512 + lane * 16;

    v8f acc0 = {};
    v8f acc1 = {};

    #pragma unroll 2
    for (int s = 0; s < KSTEPS; ++s) {
        // Shared A fragment: 32 contiguous bytes/lane (pre-swizzled), cache-hot.
        v16h a = *(const v16h*)ap;
        // 16 int8 weights per lane per tile: global_load_b128, non-temporal.
        v4u wv0 = __builtin_nontemporal_load((const v4u*)wp0);
        v4u wv1 = __builtin_nontemporal_load((const v4u*)wp1);

        union { v16h h; uint32_t u[8]; } b0, b1;
        dq4(wv0.x, &b0.u[0], &b0.u[1]);
        dq4(wv0.y, &b0.u[2], &b0.u[3]);
        dq4(wv0.z, &b0.u[4], &b0.u[5]);
        dq4(wv0.w, &b0.u[6], &b0.u[7]);
        dq4(wv1.x, &b1.u[0], &b1.u[1]);
        dq4(wv1.y, &b1.u[2], &b1.u[3]);
        dq4(wv1.z, &b1.u[4], &b1.u[5]);
        dq4(wv1.w, &b1.u[6], &b1.u[7]);

        acc0 = __builtin_amdgcn_wmma_f32_16x16x32_f16(
            false, a, false, b0.h, (short)0, acc0, false, false);
        acc1 = __builtin_amdgcn_wmma_f32_16x16x32_f16(
            false, a, false, b1.h, (short)0, acc1, false, false);

        ap  += 512;   // next 32-wide k-block of A (32 lanes * 16 halfs)
        wp0 += 32;    // next 32 weight bytes within the row
        wp1 += 32;
    }

    // C/D layout (ISA 7.12.2): VGPR r, lane L: M = r + 8*(L>=16), N = L%16.
    #pragma unroll
    for (int r = 0; r < 8; ++r) {
        lds[wave][(r + 8 * hi) * 16 + n]       = acc0[r];
        lds[wave][256 + (r + 8 * hi) * 16 + n] = acc1[r];
    }
    __syncthreads();

    // Cross-wave reduce + bias fold + per-channel scale + store.
    #pragma unroll
    for (int idx = tid; idx < NTILES * 256; idx += KSPLIT * 32) {
        float s4 = lds[0][idx] + lds[1][idx] + lds[2][idx] + lds[3][idx];
        int tile = idx >> 8;
        int rem  = idx & 255;
        int m    = rem >> 4;
        int o    = o0 + tile * 16 + (rem & 15);
        out[(size_t)m * OUT_DIM + o] = (s4 - 1152.0f * rowsum[m]) * scaler[o];
    }
}

// ---------------------------------------------------------------------------
extern "C" void kernel_launch(void* const* d_in, const int* in_sizes, int n_in,
                              void* d_out, int out_size, void* d_ws, size_t ws_size,
                              hipStream_t stream) {
    (void)in_sizes; (void)n_in; (void)out_size; (void)ws_size;

    const float*  x      = (const float*)d_in[0];
    const int8_t* w      = (const int8_t*)d_in[1];
    const float*  scaler = (const float*)d_in[2];
    float*        out    = (float*)d_out;

    _Float16* xh     = (_Float16*)d_ws;                              // 256 KB
    float*    rowsum = (float*)((char*)d_ws + (size_t)B_DIM * IN_DIM * 2);

    xa_repack_kernel<<<(B_DIM * IN_DIM) / 256, 256, 0, stream>>>(x, xh);
    xa_rowsum_kernel<<<1, B_DIM * 32, 0, stream>>>(x, rowsum);
    qmm_wmma_kernel<<<OUT_DIM / (NTILES * 16), KSPLIT * 32, 0, stream>>>(
        xh, w, scaler, rowsum, out);
}
